// temporalGNN_66932770341107
// MI455X (gfx1250) — matrix operand order repaired and checked by reading, not compiled, not run on hardware
//
#include <hip/hip_runtime.h>
#include <hip/hip_bf16.h>
#include <math.h>

typedef __attribute__((ext_vector_type(16))) __bf16 v16bf;
typedef __attribute__((ext_vector_type(8)))  __bf16 v8bf;
typedef __attribute__((ext_vector_type(8)))  float  v8f;

// ---------------------------------------------------------------- utilities
__global__ void fill_kernel(float* __restrict__ p, float v, int n) {
    int i = blockIdx.x * blockDim.x + threadIdx.x;
    if (i < n) p[i] = v;
}

__global__ void deg_kernel(const int* __restrict__ dst, float* __restrict__ deg, int E) {
    int e = blockIdx.x * blockDim.x + threadIdx.x;
    if (e < E) atomicAdd(&deg[dst[e]], 1.0f);
}

__global__ void dinv_kernel(const float* __restrict__ deg, float* __restrict__ dinv, int N) {
    int i = blockIdx.x * blockDim.x + threadIdx.x;
    if (i < N) dinv[i] = rsqrtf(deg[i] + 2.0f);
}

__global__ void cnt_kernel(const int* __restrict__ batch, float* __restrict__ cnt, int N) {
    int i = blockIdx.x * blockDim.x + threadIdx.x;
    if (i < N) atomicAdd(&cnt[batch[i]], 1.0f);
}

// ------------------------------------------------ WMMA GEMM: C[N,64] = A[N,64] @ W[64,64] (+bias)
// block = 128 threads (4 waves, wave32); each wave computes a 16x64 panel.
// A staged in LDS as bf16 row-major; W staged transposed bf16 (Wt[n][k]) so every
// fragment is two aligned 16B runs -> ds_load_b128, no per-element converts.
__global__ void gemm64_wmma_kernel(const float* __restrict__ A, int lda,
                                   const float* __restrict__ W,
                                   const float* __restrict__ bias,
                                   float* __restrict__ C, int nrows) {
    __shared__ __align__(16) __bf16 AsH[64][72];   // row stride 144B (16B aligned)
    __shared__ __align__(16) __bf16 WtH[64][72];   // transposed: WtH[n][k]
    const int tid  = threadIdx.x;
    const int wave = tid >> 5;
    const int lane = tid & 31;
    const int row0 = blockIdx.x * 64;

    // stage: convert fp32 -> bf16 once, packed pair stores
#pragma unroll
    for (int c = 0; c < 16; ++c) {
        int idx = c * 128 + tid;                 // 0..2047 element pairs
        // A tile: row = idx>>5, col pair = (idx&31)*2  (coalesced float2 loads)
        int ar = idx >> 5, ac = (idx & 31) * 2;
        int gr = row0 + ar;
        float2 av = (gr < nrows) ? *(const float2*)(A + gr * lda + ac)
                                 : float2{0.0f, 0.0f};
        AsH[ar][ac]     = (__bf16)av.x;
        AsH[ar][ac + 1] = (__bf16)av.y;
        // W: n = idx&63, k pair = (idx>>6)*2  (two coalesced loads, transposed store)
        int n = idx & 63, k2 = (idx >> 6) * 2;
        WtH[n][k2]     = (__bf16)W[k2 * 64 + n];
        WtH[n][k2 + 1] = (__bf16)W[(k2 + 1) * 64 + n];
    }
    __syncthreads();

    const int m    = lane & 15;
    const int half = lane >> 4;
    const __bf16* arow = &AsH[wave * 16 + m][0];

    v8f acc[4] = {v8f{}, v8f{}, v8f{}, v8f{}};

#pragma unroll
    for (int kt = 0; kt < 2; ++kt) {
        // A fragment (16-bit 16x32 layout): lane half selects K-octet groups
        v8bf alo = *(const v8bf*)(arow + kt * 32 + half * 16);
        v8bf ahi = *(const v8bf*)(arow + kt * 32 + half * 16 + 8);
        v16bf afrag = __builtin_shufflevector(alo, ahi,
                      0, 1, 2, 3, 4, 5, 6, 7, 8, 9, 10, 11, 12, 13, 14, 15);
#pragma unroll
        for (int nt = 0; nt < 4; ++nt) {
            // B fragment (32x16): lanes 0-15 K=0..15, lanes 16-31 K=16..31, column n
            const __bf16* brow = &WtH[nt * 16 + m][kt * 32 + half * 16];
            v8bf blo = *(const v8bf*)(brow);
            v8bf bhi = *(const v8bf*)(brow + 8);
            v16bf bfrag = __builtin_shufflevector(blo, bhi,
                          0, 1, 2, 3, 4, 5, 6, 7, 8, 9, 10, 11, 12, 13, 14, 15);
            acc[nt] = __builtin_amdgcn_wmma_f32_16x16x32_bf16(
                false, afrag, false, bfrag, (short)0, acc[nt], false, false);
        }
    }

    const int wr0 = row0 + wave * 16;       // wave-uniform guard; N is a multiple of 16
    if (wr0 < nrows) {
#pragma unroll
        for (int nt = 0; nt < 4; ++nt) {
            int col = nt * 16 + m;
            float bv = bias ? bias[col] : 0.0f;
#pragma unroll
            for (int j = 0; j < 8; ++j) {
                int r = wr0 + half * 8 + j;  // D layout: VGPR j -> M=j (lanes 0-15), M=j+8 (16-31)
                C[r * 64 + col] = acc[nt][j] + bv;
            }
        }
    }
}

// ------------------------------------------------ agg init with self-loop: agg = 2*dinv^2 * xw
__global__ void init_self_kernel(float* __restrict__ agg, const float* __restrict__ xw,
                                 const float* __restrict__ dinv, int total4) {
    int i = blockIdx.x * blockDim.x + threadIdx.x;
    if (i >= total4) return;
    int e = i * 4;
    float di = dinv[e >> 6];
    float s = 2.0f * di * di;
    float4 v = *(const float4*)(xw + e);
    float4 o = {s * v.x, s * v.y, s * v.z, s * v.w};
    *(float4*)(agg + e) = o;
}

// ------------------------------------------------ edge scatter: agg[dst] += dinv[s]*dinv[d]*xw[src]
__global__ void scatter_kernel(const int* __restrict__ src, const int* __restrict__ dst,
                               const float* __restrict__ dinv, const float* __restrict__ xw,
                               float* __restrict__ agg, int E) {
    int t  = blockIdx.x * blockDim.x + threadIdx.x;
    int e  = t >> 4;
    if (e >= E) return;
    int l4 = t & 15;
    int s = src[e], d = dst[e];
    float nrm = dinv[s] * dinv[d];
    const float4 v = *(const float4*)(xw + s * 64 + l4 * 4);
    float* ap = agg + d * 64 + l4 * 4;
    atomicAdd(ap + 0, nrm * v.x);
    atomicAdd(ap + 1, nrm * v.y);
    atomicAdd(ap + 2, nrm * v.z);
    atomicAdd(ap + 3, nrm * v.w);
}

// agg[i] = (relu?)(agg[i] + b[h]) + residual[i]   (in place; self-loop already in agg)
__global__ void combine_kernel(float* __restrict__ agg, const float* __restrict__ bias,
                               const float* __restrict__ residual, int relu, int total4) {
    int i = blockIdx.x * blockDim.x + threadIdx.x;
    if (i >= total4) return;
    int e = i * 4;
    int h = e & 63;
    float4 a = *(const float4*)(agg + e);
    const float4 bv = *(const float4*)(bias + h);
    const float4 r = *(const float4*)(residual + e);
    float4 v = {a.x + bv.x, a.y + bv.y, a.z + bv.z, a.w + bv.w};
    if (relu) {
        v.x = fmaxf(v.x, 0.0f); v.y = fmaxf(v.y, 0.0f);
        v.z = fmaxf(v.z, 0.0f); v.w = fmaxf(v.w, 0.0f);
    }
    float4 o = {v.x + r.x, v.y + r.y, v.z + r.z, v.w + r.w};
    *(float4*)(agg + e) = o;
}

__global__ void pool_kernel(const float* __restrict__ val, const int* __restrict__ batch,
                            float* __restrict__ seq, int slice, int total) {
    int i = blockIdx.x * blockDim.x + threadIdx.x;
    if (i >= total) return;
    int n = i >> 6, h = i & 63;
    atomicAdd(&seq[batch[n] * 192 + slice * 64 + h], val[i]);
}

__global__ void seq_div_kernel(float* __restrict__ seq, const float* __restrict__ cnt, int total) {
    int i = blockIdx.x * blockDim.x + threadIdx.x;
    if (i >= total) return;
    seq[i] /= fmaxf(cnt[i / 192], 1.0f);
}

// ------------------------------------------------ LSTM: one block handles 64 batch rows
__device__ __forceinline__ float sigmoidf_(float x) { return 1.0f / (1.0f + __expf(-x)); }

__global__ void lstm_kernel(const float* __restrict__ seq, const float* __restrict__ h0,
                            const float* __restrict__ c0, const float* __restrict__ Wih,
                            const float* __restrict__ Whh, const float* __restrict__ bih,
                            const float* __restrict__ bhh, float* __restrict__ hs) {
    __shared__ float hbuf[2][64][64];
    __shared__ float cbuf[64][64];
    const int tid = threadIdx.x;            // 256 threads
    const int gb0 = blockIdx.x * 64;

#pragma unroll
    for (int p = 0; p < 16; ++p) {
        int q = p * 256 + tid;              // 0..4095
        int bl = q >> 6, hh = q & 63;
        hbuf[0][bl][hh] = h0[(gb0 + bl) * 64 + hh];
        cbuf[bl][hh]    = c0[(gb0 + bl) * 64 + hh];
    }
    __syncthreads();

    int srcb = 0;
    for (int t = 0; t < 3; ++t) {
#pragma unroll
        for (int p = 0; p < 16; ++p) {
            int q = p * 256 + tid;
            int bl = q >> 6, hh = q & 63;
            float ai = bih[hh]        + bhh[hh];
            float af = bih[64 + hh]   + bhh[64 + hh];
            float ag = bih[128 + hh]  + bhh[128 + hh];
            float ao = bih[192 + hh]  + bhh[192 + hh];
            const float* xrow = seq + (gb0 + bl) * 192 + t * 64;
            const float* hrow = &hbuf[srcb][bl][0];
            for (int j = 0; j < 64; ++j) {
                float xv = xrow[j], hv = hrow[j];
                ai += xv * Wih[hh * 64 + j]         + hv * Whh[hh * 64 + j];
                af += xv * Wih[(64 + hh) * 64 + j]  + hv * Whh[(64 + hh) * 64 + j];
                ag += xv * Wih[(128 + hh) * 64 + j] + hv * Whh[(128 + hh) * 64 + j];
                ao += xv * Wih[(192 + hh) * 64 + j] + hv * Whh[(192 + hh) * 64 + j];
            }
            float cv = cbuf[bl][hh];
            float cn = sigmoidf_(af) * cv + sigmoidf_(ai) * tanhf(ag);
            float hn = sigmoidf_(ao) * tanhf(cn);
            cbuf[bl][hh] = cn;
            hbuf[1 - srcb][bl][hh] = hn;
            hs[(gb0 + bl) * 192 + t * 64 + hh] = hn;
        }
        __syncthreads();
        srcb ^= 1;
    }
}

// ------------------------------------------------ attention + head: one 64-thread block per row
__global__ void attn_head_kernel(const float* __restrict__ hs, const float* __restrict__ w0W,
                                 const float* __restrict__ w0b, const float* __restrict__ aW,
                                 const float* __restrict__ ab, const float* __restrict__ l1W,
                                 const float* __restrict__ l1b, const float* __restrict__ cov,
                                 const float* __restrict__ l2W, const float* __restrict__ l2b,
                                 float* __restrict__ out, int B) {
    __shared__ float rsh[3][64];
    __shared__ float logit[3];
    __shared__ float attn[3];
    __shared__ float pooled[64];
    __shared__ float feat[16];
    const int b = blockIdx.x, h = threadIdx.x;

#pragma unroll
    for (int t = 0; t < 3; ++t) {
        float acc = w0b[h];
        for (int j = 0; j < 64; ++j) acc += hs[b * 192 + t * 64 + j] * w0W[h * 64 + j];
        rsh[t][h] = tanhf(acc);
    }
    __syncthreads();
    if (h < 3) {
        float a = ab[0];
        for (int j = 0; j < 64; ++j) a += rsh[h][j] * aW[j];
        logit[h] = a;
    }
    __syncthreads();
    if (h == 0) {
        float mx = fmaxf(logit[0], fmaxf(logit[1], logit[2]));
        float e0 = __expf(logit[0] - mx), e1 = __expf(logit[1] - mx), e2 = __expf(logit[2] - mx);
        float inv = 1.0f / (e0 + e1 + e2);
        attn[0] = e0 * inv; attn[1] = e1 * inv; attn[2] = e2 * inv;
        out[b * 3 + 0] = attn[0]; out[b * 3 + 1] = attn[1]; out[b * 3 + 2] = attn[2];
    }
    __syncthreads();
    pooled[h] = rsh[0][h] * attn[0] + rsh[1][h] * attn[1] + rsh[2][h] * attn[2];
    __syncthreads();
    if (h < 8) {
        float a = l1b[h];
        for (int j = 0; j < 64; ++j) a += pooled[j] * l1W[h * 64 + j];
        feat[h] = fmaxf(a, 0.0f);
    } else if (h < 16) {
        feat[h] = cov[b * 8 + (h - 8)];
    }
    __syncthreads();
    if (h < 2) {
        float a = l2b[h];
        for (int j = 0; j < 16; ++j) a += feat[j] * l2W[h * 16 + j];
        out[B * 3 + b * 2 + h] = a;
    }
}

// ---------------------------------------------------------------- launcher
extern "C" void kernel_launch(void* const* d_in, const int* in_sizes, int n_in,
                              void* d_out, int out_size, void* d_ws, size_t ws_size,
                              hipStream_t stream) {
    const float* x     = (const float*)d_in[0];
    const int*   ei    = (const int*)d_in[1];
    const float* cov   = (const float*)d_in[2];
    const int*   batch = (const int*)d_in[3];
    const float* h0    = (const float*)d_in[4];
    const float* c0    = (const float*)d_in[5];
    const float* linW  = (const float*)d_in[6];
    const float* linb  = (const float*)d_in[7];
    const float* c1W   = (const float*)d_in[8];
    const float* c1b   = (const float*)d_in[9];
    const float* c2W   = (const float*)d_in[10];
    const float* c2b   = (const float*)d_in[11];
    const float* Wih   = (const float*)d_in[12];
    const float* Whh   = (const float*)d_in[13];
    const float* bih   = (const float*)d_in[14];
    const float* bhh   = (const float*)d_in[15];
    const float* w0W   = (const float*)d_in[16];
    const float* w0b   = (const float*)d_in[17];
    const float* aW    = (const float*)d_in[18];
    const float* ab    = (const float*)d_in[19];
    const float* l1W   = (const float*)d_in[20];
    const float* l1b   = (const float*)d_in[21];
    const float* l2W   = (const float*)d_in[22];
    const float* l2b   = (const float*)d_in[23];
    float* out = (float*)d_out;

    const int N = in_sizes[0] / 192;
    const int E = in_sizes[1] / 2;
    const int B = in_sizes[4] / 64;
    const int NH = N * 64;
    const int* src = ei;
    const int* dst = ei + E;

    // workspace layout (floats)
    float* ws   = (float*)d_ws;
    float* deg  = ws;            ws += N;
    float* dinv = ws;            ws += N;
    float* A0   = ws;            ws += NH;
    float* A1   = ws;            ws += NH;
    float* A2   = ws;            ws += NH;
    float* cnt  = ws;            ws += B;
    float* seq  = ws;            ws += B * 192;
    float* hs   = ws;            ws += B * 192;

    const int TB = 256;
    auto blocks = [](int n, int tb) { return (n + tb - 1) / tb; };

    // degrees + batch counts + zero pooled seq
    fill_kernel<<<blocks(N, TB), TB, 0, stream>>>(deg, 0.0f, N);
    deg_kernel<<<blocks(E, TB), TB, 0, stream>>>(dst, deg, E);
    dinv_kernel<<<blocks(N, TB), TB, 0, stream>>>(deg, dinv, N);
    fill_kernel<<<blocks(B, TB), TB, 0, stream>>>(cnt, 0.0f, B);
    cnt_kernel<<<blocks(N, TB), TB, 0, stream>>>(batch, cnt, N);
    fill_kernel<<<blocks(B * 192, TB), TB, 0, stream>>>(seq, 0.0f, B * 192);

    const int gemm_blocks = (N + 63) / 64;
    const int scat_blocks = blocks(E * 16, TB);
    const int q_blocks    = blocks(NH / 4, TB);

    for (int s = 0; s < 3; ++s) {
        // lin: A0 = x[:, s*64:(s+1)*64] @ linW + linb
        gemm64_wmma_kernel<<<gemm_blocks, 128, 0, stream>>>(x + s * 64, 192, linW, linb, A0, N);
        // conv1: xw -> A1 ; agg (init with self-loop) -> A2 ; out1 in place in A2
        gemm64_wmma_kernel<<<gemm_blocks, 128, 0, stream>>>(A0, 64, c1W, nullptr, A1, N);
        init_self_kernel<<<q_blocks, TB, 0, stream>>>(A2, A1, dinv, NH / 4);
        scatter_kernel<<<scat_blocks, TB, 0, stream>>>(src, dst, dinv, A1, A2, E);
        combine_kernel<<<q_blocks, TB, 0, stream>>>(A2, c1b, A0, 1, NH / 4);
        // conv2: xw -> A1 ; agg -> A0 ; out2 in place in A0
        gemm64_wmma_kernel<<<gemm_blocks, 128, 0, stream>>>(A2, 64, c2W, nullptr, A1, N);
        init_self_kernel<<<q_blocks, TB, 0, stream>>>(A0, A1, dinv, NH / 4);
        scatter_kernel<<<scat_blocks, TB, 0, stream>>>(src, dst, dinv, A1, A0, E);
        combine_kernel<<<q_blocks, TB, 0, stream>>>(A0, c2b, A2, 0, NH / 4);
        // global mean pool (sum; divide later)
        pool_kernel<<<blocks(NH, TB), TB, 0, stream>>>(A0, batch, seq, s, NH);
    }

    seq_div_kernel<<<blocks(B * 192, TB), TB, 0, stream>>>(seq, cnt, B * 192);
    lstm_kernel<<<B / 64, 256, 0, stream>>>(seq, h0, c0, Wih, Whh, bih, bhh, hs);
    attn_head_kernel<<<B, 64, 0, stream>>>(hs, w0W, w0b, aW, ab, l1W, l1b, cov, l2W, l2b, out, B);
}
